// GatedDeltaProductBlock_30013231464540
// MI455X (gfx1250) — compile-verified, matrix operand order checked by reading, not run on hardware
//
#include <hip/hip_runtime.h>

// ---------------------------------------------------------------------------
// Gated DeltaProduct block for MI455X (gfx1250, wave32, WMMA).
// GEMMs: bf16 WMMA (v_wmma_f32_16x16x32_bf16), f32 accumulate, with
//        double-buffered GLOBAL_LOAD_ASYNC_TO_LDS tile feed (ASYNCcnt).
// Scan:  register-resident state (128x256 f32 per (b,h)), LDS reductions.
// ---------------------------------------------------------------------------

typedef __bf16 bf16_t;
typedef __bf16 v16bf __attribute__((ext_vector_type(16)));
typedef float  v8f   __attribute__((ext_vector_type(8)));

constexpr int Bc = 2, Tc = 2048, Dc = 1024, Hc = 8, DKc = 128, DVc = 256;
constexpr int NHc = 2, CONVc = 4, INTERc = 2816;
constexpr int Mrows = Bc * Tc;            // 4096
constexpr float EPSc = 1e-6f;

__device__ __forceinline__ float sigmoidf_(float x) { return 1.0f / (1.0f + __expf(-x)); }
__device__ __forceinline__ float siluf_(float x)    { return x * sigmoidf_(x); }

// Low 32 bits of a flat LDS pointer == LDS byte offset (ISA 10.2 aperture map).
__device__ __forceinline__ unsigned ldsoff(const void* p) { return (unsigned)(size_t)p; }

// Async DMA global->LDS, tracked by ASYNCcnt (CDNA5 ISA).
__device__ __forceinline__ void async_ld_b128(unsigned lds_off, const void* g) {
    asm volatile("global_load_async_to_lds_b128 %0, %1, off"
                 :: "v"(lds_off), "v"(g) : "memory");
}
__device__ __forceinline__ void async_ld_b32(unsigned lds_off, const void* g) {
    asm volatile("global_load_async_to_lds_b32 %0, %1, off"
                 :: "v"(lds_off), "v"(g) : "memory");
}

// ------------------------------ rmsnorm (D=1024) ---------------------------
__global__ __launch_bounds__(256) void rmsnorm1024_kernel(
    const float* __restrict__ x, const float* __restrict__ w, float* __restrict__ y)
{
    __shared__ float red[256];
    const int row = blockIdx.x;
    const int tid = threadIdx.x;
    const float* xr = x + (size_t)row * 1024;
    float v[4];
    float s = 0.0f;
#pragma unroll
    for (int i = 0; i < 4; ++i) { v[i] = xr[tid + 256 * i]; s += v[i] * v[i]; }
    red[tid] = s;
    __syncthreads();
    for (int off = 128; off > 0; off >>= 1) {
        if (tid < off) red[tid] += red[tid + off];
        __syncthreads();
    }
    const float inv = rsqrtf(red[0] * (1.0f / 1024.0f) + EPSc);
    float* yr = y + (size_t)row * 1024;
#pragma unroll
    for (int i = 0; i < 4; ++i) yr[tid + 256 * i] = v[i] * inv * w[tid + 256 * i];
}

// ------------------------- bf16 WMMA GEMM 128x128x32 -----------------------
// C[M,N] = A[M,K] @ W[K,N] (+ addend). fp32 tiles stream into LDS via async
// DMA (double buffered); f32->bf16 conversion happens at fragment build.
// 256 threads = 8 waves; wave (wm,wn) owns a 32x64 sub-tile = 2x4 WMMA tiles.
#define BM 128
#define BN 128
#define BK 32

__global__ __launch_bounds__(256) void gemm_bf16_wmma_kernel(
    const float* __restrict__ A, const float* __restrict__ W,
    const float* __restrict__ addend, float* __restrict__ C,
    int Mi, int Ni, int Ki)
{
    __shared__ alignas(16) float As[2][BM][BK];   // row-major M x K tile (f32)
    __shared__ alignas(16) float Wt[2][BN][BK];   // transposed [n][k]   (f32)

    const int tid  = threadIdx.x;
    const int wave = tid >> 5;
    const int lane = tid & 31;
    const int wm   = wave >> 1;       // 0..3
    const int wn   = wave & 1;        // 0..1
    const int hlf  = lane >> 4;       // lane half (ISA 7.12.2 fragment layout)
    const int l16  = lane & 15;
    const int brow = blockIdx.y * BM;
    const int bcol = blockIdx.x * BN;

    // per-thread DMA coordinates: 16 contiguous f32 of the A tile (4 x b128)
    // and 16 contiguous f32 of one W row, scattered into [n][k] (16 x b32).
    const int aRow = (tid * 16) >> 5;      // 0..127
    const int aCol = (tid * 16) & 31;      // 0 or 16
    const int wK   = (tid * 16) >> 7;      // 0..31
    const int wN   = (tid * 16) & 127;     // 0,16,...,112

    auto issue_tile = [&](int buf, int k0) {
        const float* ga = &A[(size_t)(brow + aRow) * Ki + k0 + aCol];
        const unsigned la = ldsoff(&As[buf][aRow][aCol]);
#pragma unroll
        for (int j = 0; j < 4; ++j)
            async_ld_b128(la + j * 16, (const char*)ga + j * 16);
        const float* gw = &W[(size_t)(k0 + wK) * Ni + bcol + wN];
#pragma unroll
        for (int j = 0; j < 16; ++j)   // coalesced global, transposed LDS
            async_ld_b32(ldsoff(&Wt[buf][wN + j][wK]), gw + j);
    };

    v8f acc[2][4];
#pragma unroll
    for (int mt = 0; mt < 2; ++mt)
#pragma unroll
        for (int nt = 0; nt < 4; ++nt)
#pragma unroll
            for (int r = 0; r < 8; ++r) acc[mt][nt][r] = 0.0f;

    issue_tile(0, 0);                       // prologue: 20 async ops in flight

    int buf = 0;
    for (int k0 = 0; k0 < Ki; k0 += BK) {
        if (k0 + BK < Ki) {
            issue_tile(buf ^ 1, k0 + BK);   // 40 in flight; wait <=20 ->
            asm volatile("s_wait_asynccnt 0x14" ::: "memory");  // current tile done
        } else {
            asm volatile("s_wait_asynccnt 0x0" ::: "memory");
        }
        __syncthreads();

        // A fragments: lanes 0-15 hold K {0..7,16..23}, lanes 16-31 {8..15,24..31}
        v16bf af[2];
#pragma unroll
        for (int mt = 0; mt < 2; ++mt) {
            const int row = wm * 32 + mt * 16 + l16;
            const float* ar = &As[buf][row][hlf * 8];
#pragma unroll
            for (int i = 0; i < 8; ++i) {
                af[mt][i]     = (bf16_t)ar[i];
                af[mt][8 + i] = (bf16_t)ar[16 + i];
            }
        }
#pragma unroll
        for (int nt = 0; nt < 4; ++nt) {
            v16bf bfrag;
            const int col = wn * 64 + nt * 16 + l16;
            const float* br = &Wt[buf][col][hlf * 8];
#pragma unroll
            for (int i = 0; i < 8; ++i) {
                bfrag[i]     = (bf16_t)br[i];
                bfrag[8 + i] = (bf16_t)br[16 + i];
            }
#pragma unroll
            for (int mt = 0; mt < 2; ++mt) {
                acc[mt][nt] = __builtin_amdgcn_wmma_f32_16x16x32_bf16(
                    false, af[mt], false, bfrag, (short)0, acc[mt][nt], false, false);
            }
        }
        __syncthreads();
        buf ^= 1;
    }

    // C layout: VGPR r -> row r (lanes 0-15) / row r+8 (lanes 16-31), col = lane%16
#pragma unroll
    for (int mt = 0; mt < 2; ++mt)
#pragma unroll
        for (int nt = 0; nt < 4; ++nt)
#pragma unroll
            for (int r = 0; r < 8; ++r) {
                const int row = brow + wm * 32 + mt * 16 + hlf * 8 + r;
                const int col = bcol + wn * 64 + nt * 16 + l16;
                float v = acc[mt][nt][r];
                if (addend) v += addend[(size_t)row * Ni + col];
                C[(size_t)row * Ni + col] = v;
            }
}

// ------------------ small projections: beta (sigmoid), g ------------------
__global__ __launch_bounds__(256) void smallproj_kernel(
    const float* __restrict__ xn, const float* __restrict__ Wb,
    const float* __restrict__ Wa, const float* __restrict__ A_log,
    const float* __restrict__ dt_bias, float* __restrict__ beta,
    float* __restrict__ g)
{
    const int idx = blockIdx.x * blockDim.x + threadIdx.x;
    if (idx >= Mrows * 24) return;
    const int row = idx / 24, c = idx % 24;
    const float* xr = xn + (size_t)row * Dc;
    float s = 0.0f;
    if (c < 16) {
        for (int k = 0; k < Dc; ++k) s += xr[k] * Wb[k * 16 + c];
        beta[row * 16 + c] = sigmoidf_(s);
    } else {
        const int h = c - 16;
        for (int k = 0; k < Dc; ++k) s += xr[k] * Wa[k * 8 + h];
        const float z = s + dt_bias[h];
        const float sp = (z > 20.0f) ? z : log1pf(__expf(z));
        g[row * 8 + h] = -__expf(A_log[h]) * sp;
    }
}

// -------------------- causal depthwise conv (K=4) + SiLU -------------------
__global__ __launch_bounds__(256) void conv_silu_kernel(
    const float* __restrict__ x, const float* __restrict__ w,
    float* __restrict__ y, int C)
{
    const long long idx = (long long)blockIdx.x * blockDim.x + threadIdx.x;
    const long long total = (long long)Mrows * C;
    if (idx >= total) return;
    const int c = (int)(idx % C);
    const long long rt = idx / C;           // b*T + t
    const int t = (int)(rt % Tc);
    float acc = 0.0f;
#pragma unroll
    for (int i = 0; i < CONVc; ++i) {
        const int ts = t - (CONVc - 1) + i;
        if (ts >= 0) acc += x[(rt + (ts - t)) * C + c] * w[c * CONVc + i];
    }
    y[idx] = siluf_(acc);
}

// ---------------------------- l2norm over 128 ------------------------------
__global__ __launch_bounds__(128) void l2norm128_kernel(float* __restrict__ x)
{
    __shared__ float red[128];
    const int tid = threadIdx.x;
    float* p = x + (size_t)blockIdx.x * 128;
    const float v = p[tid];
    red[tid] = v * v;
    __syncthreads();
    for (int off = 64; off > 0; off >>= 1) {
        if (tid < off) red[tid] += red[tid + off];
        __syncthreads();
    }
    p[tid] = v * rsqrtf(red[0] + 1e-6f);
}

// --------------------- sequential gated delta-rule scan --------------------
// One block per (b,h). State S[128][256] in registers: thread (kg,vg) owns
// k rows kg*8..+8 and v cols vg*16..+16 (128 f32 regs/thread).
__global__ __launch_bounds__(256) void scan_kernel(
    const float* __restrict__ q, const float* __restrict__ k,
    const float* __restrict__ v, const float* __restrict__ beta,
    const float* __restrict__ g, float* __restrict__ o)
{
    __shared__ float shq[128], shk[2][128], shv[2][256];
    __shared__ float partial[16][256];
    __shared__ float red[256];
    __shared__ float shb[2], shg;

    const int b = blockIdx.x / Hc, h = blockIdx.x % Hc;
    const int tid = threadIdx.x;
    const int kg = tid >> 4, vg = tid & 15;

    float S[8][16];
#pragma unroll
    for (int i = 0; i < 8; ++i)
#pragma unroll
        for (int j = 0; j < 16; ++j) S[i][j] = 0.0f;

    for (int t = 0; t < Tc; ++t) {
        const size_t row = (size_t)b * Tc + t;
        if (tid < 128) {
            shq[tid]    = q[(row * Hc + h) * DKc + tid];
            shk[0][tid] = k[((row * NHc + 0) * Hc + h) * DKc + tid];
            shk[1][tid] = k[((row * NHc + 1) * Hc + h) * DKc + tid];
        }
        shv[0][tid] = v[((row * NHc + 0) * Hc + h) * DVc + tid];
        shv[1][tid] = v[((row * NHc + 1) * Hc + h) * DVc + tid];
        if (tid < 2)  shb[tid] = beta[(row * Hc + h) * NHc + tid];
        if (tid == 0) shg = g[row * Hc + h];
        __syncthreads();

        const float dec = __expf(shg);
#pragma unroll
        for (int kk = 0; kk < 8; ++kk)
#pragma unroll
            for (int vv = 0; vv < 16; ++vv) S[kk][vv] *= dec;

        for (int j = 0; j < NHc; ++j) {
            // kS[v] = sum_k k[k] * S[k][v]  (partial over this thread's 8 k's)
#pragma unroll
            for (int vv = 0; vv < 16; ++vv) {
                float p = 0.0f;
#pragma unroll
                for (int kk = 0; kk < 8; ++kk) p += shk[j][kg * 8 + kk] * S[kk][vv];
                partial[kg][vg * 16 + vv] = p;
            }
            __syncthreads();
            float s = 0.0f;
#pragma unroll
            for (int gg = 0; gg < 16; ++gg) s += partial[gg][tid];
            red[tid] = s;
            __syncthreads();
            // S += beta_j * k ⊗ (v_j - kS)
            const float bj = shb[j];
#pragma unroll
            for (int vv = 0; vv < 16; ++vv) {
                const int vi = vg * 16 + vv;
                const float dvv = bj * (shv[j][vi] - red[vi]);
#pragma unroll
                for (int kk = 0; kk < 8; ++kk) S[kk][vv] += shk[j][kg * 8 + kk] * dvv;
            }
        }

        // o[v] = sum_k q[k] * S[k][v]
#pragma unroll
        for (int vv = 0; vv < 16; ++vv) {
            float p = 0.0f;
#pragma unroll
            for (int kk = 0; kk < 8; ++kk) p += shq[kg * 8 + kk] * S[kk][vv];
            partial[kg][vg * 16 + vv] = p;
        }
        __syncthreads();
        float s = 0.0f;
#pragma unroll
        for (int gg = 0; gg < 16; ++gg) s += partial[gg][tid];
        o[(row * Hc + h) * DVc + tid] = s;
        // top-of-loop __syncthreads() separates this read of `partial`
        // from the next step's writes
    }
}

// ----------------- o = rmsnorm(o)*w * silu(gate), in place -----------------
__global__ __launch_bounds__(256) void ogate_kernel(
    float* __restrict__ o, const float* __restrict__ gate, const float* __restrict__ w)
{
    __shared__ float red[256];
    const size_t base = (size_t)blockIdx.x * 256;
    const int tid = threadIdx.x;
    const float val = o[base + tid];
    red[tid] = val * val;
    __syncthreads();
    for (int off = 128; off > 0; off >>= 1) {
        if (tid < off) red[tid] += red[tid + off];
        __syncthreads();
    }
    const float inv = rsqrtf(red[0] * (1.0f / 256.0f) + EPSc);
    o[base + tid] = val * inv * w[tid] * siluf_(gate[base + tid]);
}

// ------------------------------ SwiGLU gate --------------------------------
__global__ __launch_bounds__(256) void mlpact_kernel(
    const float* __restrict__ gu, float* __restrict__ a)
{
    const long long idx = (long long)blockIdx.x * blockDim.x + threadIdx.x;
    if (idx >= (long long)Mrows * INTERc) return;
    const long long row = idx / INTERc;
    const int i = (int)(idx % INTERc);
    const float* r = gu + row * (2 * INTERc);
    a[idx] = siluf_(r[i]) * r[i + INTERc];
}

// ---------------------------------------------------------------------------
extern "C" void kernel_launch(void* const* d_in, const int* in_sizes, int n_in,
                              void* d_out, int out_size, void* d_ws, size_t ws_size,
                              hipStream_t stream)
{
    const float* hs    = (const float*)d_in[0];
    const float* anw   = (const float*)d_in[1];
    const float* Wq    = (const float*)d_in[2];
    const float* Wk    = (const float*)d_in[3];
    const float* Wv    = (const float*)d_in[4];
    const float* Wb    = (const float*)d_in[5];
    const float* Wa    = (const float*)d_in[6];
    const float* A_log = (const float*)d_in[7];
    const float* dtb   = (const float*)d_in[8];
    const float* cqw   = (const float*)d_in[9];
    const float* ckw   = (const float*)d_in[10];
    const float* cvw   = (const float*)d_in[11];
    const float* Wg    = (const float*)d_in[12];
    const float* onw   = (const float*)d_in[13];
    const float* Wo    = (const float*)d_in[14];
    const float* mnw   = (const float*)d_in[15];
    const float* Wgate = (const float*)d_in[16];
    const float* Wdown = (const float*)d_in[17];
    float* out = (float*)d_out;

    const size_t M = Mrows;
    float* ws    = (float*)d_ws;
    float* xn    = ws;                    // M*1024  rmsnorm(x)
    float* qpre  = xn    + M * 1024;      // M*1024  x@Wq (pre-conv)
    float* kpre  = qpre  + M * 1024;      // M*2048
    float* vpre  = kpre  + M * 2048;      // M*4096
    float* qb    = vpre  + M * 4096;      // M*1024  post conv/silu/l2norm
    float* kb    = qb    + M * 1024;      // M*2048
    float* vb    = kb    + M * 2048;      // M*4096
    float* betab = vb    + M * 4096;      // M*16
    float* gb    = betab + M * 16;        // M*8
    float* gateb = gb    + M * 8;         // M*2048  x@Wg
    float* ob    = gateb + M * 2048;      // M*2048  scan output / gated o
    float* hid   = ob    + M * 2048;      // M*1024  post-attn residual
    float* yn    = hid   + M * 1024;      // M*1024  mlp rmsnorm
    float* gu    = qpre;                  // reuse pre-conv region: M*5632
    float* act   = vb;                    // reuse v region: M*2816

    // 1) attn rmsnorm
    rmsnorm1024_kernel<<<(int)M, 256, 0, stream>>>(hs, anw, xn);

    // 2) projections (bf16 WMMA GEMMs, async-to-LDS tile feed)
    gemm_bf16_wmma_kernel<<<dim3(1024 / BN, M / BM), 256, 0, stream>>>(xn, Wq, nullptr, qpre, M, 1024, 1024);
    gemm_bf16_wmma_kernel<<<dim3(2048 / BN, M / BM), 256, 0, stream>>>(xn, Wk, nullptr, kpre, M, 2048, 1024);
    gemm_bf16_wmma_kernel<<<dim3(4096 / BN, M / BM), 256, 0, stream>>>(xn, Wv, nullptr, vpre, M, 4096, 1024);
    gemm_bf16_wmma_kernel<<<dim3(2048 / BN, M / BM), 256, 0, stream>>>(xn, Wg, nullptr, gateb, M, 2048, 1024);
    smallproj_kernel<<<(int)((M * 24 + 255) / 256), 256, 0, stream>>>(xn, Wb, Wa, A_log, dtb, betab, gb);

    // 3) causal conv + silu
    conv_silu_kernel<<<(int)((M * 1024 + 255) / 256), 256, 0, stream>>>(qpre, cqw, qb, 1024);
    conv_silu_kernel<<<(int)((M * 2048 + 255) / 256), 256, 0, stream>>>(kpre, ckw, kb, 2048);
    conv_silu_kernel<<<(int)((M * 4096 + 255) / 256), 256, 0, stream>>>(vpre, cvw, vb, 4096);

    // 4) l2norm on q (M*H groups) and k (M*NH*H groups)
    l2norm128_kernel<<<(int)(M * 8), 128, 0, stream>>>(qb);
    l2norm128_kernel<<<(int)(M * 16), 128, 0, stream>>>(kb);

    // 5) sequential scan, one block per (b,h)
    scan_kernel<<<Bc * Hc, 256, 0, stream>>>(qb, kb, vb, betab, gb, ob);

    // 6) gated output norm (in place on ob)
    ogate_kernel<<<(int)(M * 8), 256, 0, stream>>>(ob, gateb, onw);

    // 7) hidden = residual + o @ Wo
    gemm_bf16_wmma_kernel<<<dim3(1024 / BN, M / BM), 256, 0, stream>>>(ob, Wo, hs, hid, M, 1024, 2048);

    // 8) MLP
    rmsnorm1024_kernel<<<(int)M, 256, 0, stream>>>(hid, mnw, yn);
    gemm_bf16_wmma_kernel<<<dim3(5632 / BN, M / BM), 256, 0, stream>>>(yn, Wgate, nullptr, gu, M, 5632, 1024);
    mlpact_kernel<<<(int)((M * (long long)INTERc + 255) / 256), 256, 0, stream>>>(gu, act);
    gemm_bf16_wmma_kernel<<<dim3(1024 / BN, M / BM), 256, 0, stream>>>(act, Wdown, hid, out, M, 1024, INTERc);
}
